// SegmentedSAE_12730283065912
// MI455X (gfx1250) — matrix operand order repaired and checked by reading, not compiled, not run on hardware
//
#include <hip/hip_runtime.h>
#include <hip/hip_bf16.h>
#include <math.h>

// ---------------- problem constants ----------------
#define BATCH   2048
#define DM      256
#define NF      32768
#define NBLK    8
#define MT      64      // batch rows per workgroup tile
#define CH      128     // feature chunk width
#define FG      2048    // features per workgroup group (phase 1)
#define NG      (NF / FG)
#define LOG01   (-2.3025850929940457f)

// LDS strides (bf16 elements), padded so every 16B fragment read stays aligned
#define XNS     264     // xn tile   [64][256] stride
#define WES     264     // W_enc^T   [128][256] stride
#define WDS     136     // W_dec^T   [256][128] stride
#define AS      136     // acts tile [64][128] stride

// LDS map: double-buffered weight tiles (ping-pong DMA), 326,656 B <= 320 KB
#define SM_XN     0               // 64*264*2   = 33792
#define SM_ACTS   33792           // 64*136*2   = 17408
#define SM_BENC   51200           // 2*128*4    = 1024
#define SM_WENC0  52224           // 128*264*2  = 67584
#define WENC_D    67584           // buffer delta
#define SM_WDEC0  187392          // 256*136*2  = 69632
#define WDEC_D    69632           // buffer delta
#define SMEM_BYTES 326656

#define NP (NBLK * BATCH * DM)    // per-block partial decode sums

typedef __attribute__((ext_vector_type(16))) __bf16 v16bf;
typedef __attribute__((ext_vector_type(8)))  float  v8f;

struct Frag { union { v16bf v; uint4 u[2]; }; };

static __device__ inline v8f wmma_bf16(const Frag& a, const Frag& b, v8f c) {
  return __builtin_amdgcn_wmma_f32_16x16x32_bf16(false, a.v, false, b.v,
                                                 (short)0, c, false, false);
}

// B fragment: 8 bf16 at +0 and 8 bf16 at +16 elements
static __device__ inline void ld_fragB(Frag& f, const __bf16* p) {
  f.u[0] = *(const uint4*)p;
  f.u[1] = *(const uint4*)(p + 16);
}
// A fragment: 8 bf16 at +0 and 8 bf16 at +16 elements
static __device__ inline void ld_fragA(Frag& f, const __bf16* p) {
  f.u[0] = *(const uint4*)p;
  f.u[1] = *(const uint4*)(p + 16);
}

static __device__ inline float wave_sum(float v) {
  #pragma unroll
  for (int o = 16; o > 0; o >>= 1) v += __shfl_down(v, o, 32);
  return v;
}

// async DMA: global -> LDS, 16 bytes per lane, tracked by ASYNCcnt
static __device__ inline void async_b128(unsigned lds_off, const void* gptr) {
  asm volatile("global_load_async_to_lds_b128 %0, %1, off"
               :: "v"(lds_off), "v"((unsigned long long)(uintptr_t)gptr)
               : "memory");
}
static __device__ inline void wait_async0() {
  asm volatile("s_wait_asynccnt 0x0" ::: "memory");
}

// issue one chunk's weight-tile DMA (32 async b128 per thread) + bias loads
static __device__ inline void stage_chunk(__bf16* We, __bf16* Wd, float* bS,
                                          const __bf16* wencTg, const __bf16* wdecTg,
                                          const float* benc, int cb,
                                          int w, int lane, int khalf, int tid) {
  #pragma unroll 4
  for (int it = 0; it < 16; ++it) {                 // W_enc^T: 128 rows x 512B
    int row = it * 8 + w;
    unsigned lds = (unsigned)(uintptr_t)(We + row * WES + lane * 8);
    async_b128(lds, wencTg + (size_t)(cb + row) * DM + lane * 8);
  }
  #pragma unroll 4
  for (int it = 0; it < 16; ++it) {                 // W_dec^T: 256 rows x 256B
    int row = it * 16 + w * 2 + khalf;
    unsigned lds = (unsigned)(uintptr_t)(Wd + row * WDS + (lane & 15) * 8);
    async_b128(lds, wdecTg + (size_t)row * NF + cb + (lane & 15) * 8);
  }
  if (tid < CH) bS[tid] = benc[cb + tid];
}

// ---------------- small prep kernels ----------------
__global__ void sae_zero(float* wsf) {
  int t = threadIdx.x;
  if (t < 24) wsf[t] = 0.0f;     // [0]=sumNorm [1]=scale [2..19]=accums
}

__global__ void sae_zeroP(float* __restrict__ P) {
  for (size_t i = (size_t)blockIdx.x * blockDim.x + threadIdx.x; i < NP;
       i += (size_t)gridDim.x * blockDim.x) P[i] = 0.0f;
}

// one-time bf16, B-major copies of the weights:
//   wencTg[f*256 + k] = bf16(W_enc[k][f])   (encoder B: column f, contiguous K)
//   wdecTg[n*NF  + f] = bf16(W_dec[f][n])   (decoder B: column n, contiguous K=f)
__global__ void sae_cvt(const float* __restrict__ Wenc, const float* __restrict__ Wdec,
                        __bf16* __restrict__ wencTg, __bf16* __restrict__ wdecTg) {
  const size_t N = (size_t)NF * DM;
  const size_t stride = (size_t)gridDim.x * blockDim.x;
  for (size_t i = (size_t)blockIdx.x * blockDim.x + threadIdx.x; i < N; i += stride) {
    size_t k = i >> 15, f = i & (NF - 1);          // i = k*NF + f : read coalesced
    wencTg[f * DM + k] = (__bf16)Wenc[i];
  }
  for (size_t i = (size_t)blockIdx.x * blockDim.x + threadIdx.x; i < N; i += stride) {
    size_t f = i >> 8, n = i & (DM - 1);           // i = f*DM + n : read coalesced
    wdecTg[n * NF + f] = (__bf16)Wdec[i];
  }
}

// mean row L2 norm of x (2048 x 256): grid 64 x 256
__global__ void sae_rownorm(const float* __restrict__ x, float* __restrict__ sumNorm) {
  int w = threadIdx.x >> 5, lane = threadIdx.x & 31;
  float acc = 0.0f;
  #pragma unroll
  for (int i = 0; i < 4; ++i) {
    int row = blockIdx.x * 32 + w * 4 + i;
    float s = 0.0f;
    for (int c = lane; c < DM; c += 32) { float v = x[(size_t)row * DM + c]; s = fmaf(v, v, s); }
    s = wave_sum(s);
    if (lane == 0) acc += sqrtf(s);
  }
  if (lane == 0) atomicAdd(sumNorm, acc);
}

__global__ void sae_scale(const float* sumNorm, const float* ravg, float* wsf) {
  if (threadIdx.x == 0) {
    float meanNorm = sumNorm[0] * (1.0f / (float)BATCH);
    float ra = 0.99f * ravg[0] + 0.01f * meanNorm;
    wsf[1] = 16.0f / ra;                           // sqrt(256)/ra
  }
}

// W_dec row norms (32768 x 256): grid 512 x 256
__global__ void sae_wdnorm(const float* __restrict__ Wdec, float* __restrict__ wdn) {
  int w = threadIdx.x >> 5, lane = threadIdx.x & 31;
  #pragma unroll
  for (int i = 0; i < 8; ++i) {
    int row = blockIdx.x * 64 + w * 8 + i;
    float s = 0.0f;
    for (int c = lane; c < DM; c += 32) { float v = Wdec[(size_t)row * DM + c]; s = fmaf(v, v, s); }
    s = wave_sum(s);
    if (lane == 0) wdn[row] = sqrtf(s);
  }
}

// seg (nondecreasing, 0..7) -> block end indices bnd[k]
__global__ void sae_bounds(const int* __restrict__ seg, int* __restrict__ bnd) {
  int t = threadIdx.x;
  if (t == 0) {
    for (int k = 0; k < seg[0]; ++k) bnd[k] = 0;
    for (int k = seg[NF - 1]; k < NBLK; ++k) bnd[k] = NF;
  }
  for (int i = t; i < NF - 1; i += 256) {
    int a = seg[i], b = seg[i + 1];
    for (int k = a; k < b; ++k) bnd[k] = i + 1;
  }
}

// ---------------- phase 1: fused encoder + per-block decode partials ----------------
// grid (32 row-tiles, 16 feature-groups) x 256 threads
__global__ __launch_bounds__(256)
void sae_phase1(const float* __restrict__ x,     const __bf16* __restrict__ wencTg,
                const float* __restrict__ benc,  const __bf16* __restrict__ wdecTg,
                const float* __restrict__ wdn,   const int*    __restrict__ bnd,
                const float* __restrict__ wsf,   float* __restrict__ accum,
                float* __restrict__ P)
{
  extern __shared__ __align__(16) char smem[];
  __bf16* xnT   = (__bf16*)(smem + SM_XN);
  __bf16* actsT = (__bf16*)(smem + SM_ACTS);
  float*  bencD = (float*) (smem + SM_BENC);

  const int tid  = threadIdx.x;
  const int lane = tid & 31;
  const int w    = tid >> 5;
  const int rowBase = blockIdx.x * MT;
  const int g0 = blockIdx.y * FG, g1 = g0 + FG;
  const float scale = wsf[1];

  for (int i = tid; i < MT * DM; i += 256) {
    int m = i >> 8, k = i & 255;
    xnT[m * XNS + k] = (__bf16)(x[(size_t)(rowBase + m) * DM + k] * scale);
  }

  const int Mt    = w >> 1;
  const int encN0 = (w & 1) * 4;
  const int decN0 = (w & 1) * 8;
  const int mrow  = Mt * 16 + (lane & 15);
  const int khalf = lane >> 4;

  for (int k = 0; k < NBLK; ++k) {
    int bs = (k == 0) ? 0 : bnd[k - 1];
    int be = bnd[k];
    bs = bs > g0 ? bs : g0;
    be = be < g1 ? be : g1;
    if (bs >= be) continue;

    v8f cum[8];
    #pragma unroll
    for (int j = 0; j < 8; ++j)
      #pragma unroll
      for (int e = 0; e < 8; ++e) cum[j][e] = 0.0f;

    const int cb0 = bs & ~(CH - 1);
    const int nch = (be - cb0 + CH - 1) / CH;

    // segment prologue: fence previous segment's readers, then kick off DMA 0
    __syncthreads();
    stage_chunk((__bf16*)(smem + SM_WENC0), (__bf16*)(smem + SM_WDEC0), bencD,
                wencTg, wdecTg, benc, cb0, w, lane, khalf, tid);

    for (int ci = 0; ci < nch; ++ci) {
      const int cb  = cb0 + ci * CH;
      const int cur = ci & 1;
      const int nxt = cur ^ 1;

      wait_async0();      // buf[cur]'s 32 loads are the only outstanding ones
      __syncthreads();    // publish buf[cur]; fences buf[cur^1] readers (ci-1)

      if (ci + 1 < nch)   // DMA for next chunk overlaps this chunk's WMMAs
        stage_chunk((__bf16*)(smem + SM_WENC0 + nxt * WENC_D),
                    (__bf16*)(smem + SM_WDEC0 + nxt * WDEC_D),
                    bencD + nxt * CH,
                    wencTg, wdecTg, benc, cb + CH, w, lane, khalf, tid);

      const __bf16* We = (const __bf16*)(smem + SM_WENC0 + cur * WENC_D);
      const __bf16* Wd = (const __bf16*)(smem + SM_WDEC0 + cur * WDEC_D);
      const float*  bS = bencD + cur * CH;

      // ---- encoder GEMM: acts[64x128] = xn @ We ; wave does 4 N-tiles ----
      // explicit one-ahead software pipeline on A (over kk) and B (over j)
      v8f acc[4];
      #pragma unroll
      for (int j = 0; j < 4; ++j)
        #pragma unroll
        for (int e = 0; e < 8; ++e) acc[j][e] = 0.0f;

      Frag a;
      ld_fragA(a, xnT + mrow * XNS + khalf * 8);
      #pragma unroll
      for (int kk = 0; kk < 8; ++kk) {
        Frag an;
        if (kk + 1 < 8)
          ld_fragA(an, xnT + mrow * XNS + (kk + 1) * 32 + khalf * 8);
        Frag b[4];
        ld_fragB(b[0], We + (encN0 * 16 + (lane & 15)) * WES + kk * 32 + khalf * 8);
        #pragma unroll
        for (int j = 0; j < 4; ++j) {
          if (j + 1 < 4)
            ld_fragB(b[j + 1], We + ((encN0 + j + 1) * 16 + (lane & 15)) * WES
                                  + kk * 32 + khalf * 8);
          acc[j] = wmma_bf16(a, b[j], acc[j]);
        }
        a = an;
      }

      // ---- bias + relu + segment mask + l1/l0 + bf16 acts to LDS ----
      float l1loc = 0.0f, l0loc = 0.0f;
      #pragma unroll
      for (int j = 0; j < 4; ++j) {
        int nloc = (encN0 + j) * 16 + (lane & 15);
        int f    = cb + nloc;
        bool inR = (f >= bs) && (f < be);
        float wn   = wdn[f];
        float bias = bS[nloc];
        #pragma unroll
        for (int r = 0; r < 8; ++r) {
          float av = fmaxf(acc[j][r] + bias, 0.0f);
          if (!inR) av = 0.0f;
          if (inR) {
            l1loc += __logf(fmaf(av, wn, 0.1f)) - LOG01;
            if (av > 0.0f) l0loc += 1.0f;
          }
          int mr = Mt * 16 + r + 8 * khalf;
          actsT[mr * AS + nloc] = (__bf16)av;
        }
      }
      l1loc = wave_sum(l1loc);
      l0loc = wave_sum(l0loc);
      if (lane == 0) { atomicAdd(&accum[k], l1loc); atomicAdd(&accum[16], l0loc); }

      __syncthreads();   // acts tile complete for all waves

      // ---- decoder GEMM: cum[64x256] += acts @ Wd ; one-ahead pipeline ----
      Frag ad;
      ld_fragA(ad, actsT + mrow * AS + khalf * 8);
      #pragma unroll
      for (int kk = 0; kk < 4; ++kk) {
        Frag an;
        if (kk + 1 < 4)
          ld_fragA(an, actsT + mrow * AS + (kk + 1) * 32 + khalf * 8);
        Frag b[8];
        ld_fragB(b[0], Wd + (decN0 * 16 + (lane & 15)) * WDS + kk * 32 + khalf * 8);
        #pragma unroll
        for (int j = 0; j < 8; ++j) {
          if (j + 1 < 8)
            ld_fragB(b[j + 1], Wd + ((decN0 + j + 1) * 16 + (lane & 15)) * WDS
                                  + kk * 32 + khalf * 8);
          cum[j] = wmma_bf16(ad, b[j], cum[j]);
        }
        ad = an;
      }
    } // chunk loop

    // ---- flush this block's partial decode into P[k] ----
    #pragma unroll
    for (int j = 0; j < 8; ++j) {
      int n = (decN0 + j) * 16 + (lane & 15);
      #pragma unroll
      for (int r = 0; r < 8; ++r) {
        int mr = rowBase + Mt * 16 + r + 8 * khalf;
        atomicAdd(&P[((size_t)k * BATCH + mr) * DM + n], cum[j][r]);
      }
    }
  } // block loop
}

// ---------------- phase 2: cumsum over blocks + error checkpoints ----------------
__global__ void sae_err(const float* __restrict__ x, const float* __restrict__ bdec,
                        const float* __restrict__ wsf, const float* __restrict__ P,
                        float* __restrict__ accum)
{
  int m = blockIdx.x, n = threadIdx.x, lane = threadIdx.x & 31;
  float xv = x[(size_t)m * DM + n] * wsf[1];      // f32 xn (full precision)
  float bd = bdec[n];
  float cum = 0.0f;
  #pragma unroll
  for (int k = 0; k < NBLK; ++k) {
    cum += P[((size_t)k * BATCH + m) * DM + n];
    float d = cum + bd - xv;
    float e = wave_sum(d * d);
    if (lane == 0) atomicAdd(&accum[8 + k], e);
  }
}

// ---------------- final scalar combine ----------------
__global__ void sae_final(const float* __restrict__ accum, const float* __restrict__ bw,
                          const float* __restrict__ l1scale, float* __restrict__ out)
{
  if (threadIdx.x == 0 && blockIdx.x == 0) {
    const float invB = 1.0f / (float)BATCH;
    float c = 0.0f, l1 = 0.0f, mse = 0.0f;
    for (int k = 0; k < NBLK; ++k) {
      c   += accum[k] * invB;
      l1  += bw[k] * c;
      mse += bw[k] * (accum[8 + k] * invB);
    }
    l1 *= 0.125f; mse *= 0.125f;
    float avg_l0 = accum[16] * invB;
    float s = (avg_l0 < 100.0f) ? l1scale[0] * (1.0f - 3e-4f)
                                : l1scale[0] * (1.0f + 3e-4f);
    out[0] = mse + s * l1;
  }
}

// ---------------- launcher ----------------
extern "C" void kernel_launch(void* const* d_in, const int* in_sizes, int n_in,
                              void* d_out, int out_size, void* d_ws, size_t ws_size,
                              hipStream_t stream) {
  const float* x     = (const float*)d_in[0];
  const float* Wenc  = (const float*)d_in[1];
  const float* benc  = (const float*)d_in[2];
  const float* Wdec  = (const float*)d_in[3];
  const float* bdec  = (const float*)d_in[4];
  const float* bw    = (const float*)d_in[5];
  const float* ravg  = (const float*)d_in[6];
  const float* l1sc  = (const float*)d_in[7];
  const int*   seg   = (const int*)  d_in[8];

  float*  wsf    = (float*)d_ws;
  float*  accum  = wsf + 2;                    // 17 floats of accumulators
  int*    bnd    = (int*)(wsf + 24);           // 8 block end indices
  float*  wdn    = wsf + 64;                   // 32768 W_dec row norms
  float*  P      = wsf + 64 + NF;              // 8*2048*256 per-block partials
  __bf16* wencTg = (__bf16*)(P + NP);          // bf16 W_enc^T [NF][DM]
  __bf16* wdecTg = wencTg + (size_t)NF * DM;   // bf16 W_dec^T [DM][NF]
  float*  out    = (float*)d_out;

  (void)hipFuncSetAttribute((const void*)sae_phase1,
                            hipFuncAttributeMaxDynamicSharedMemorySize, SMEM_BYTES);

  sae_zero   <<<1,    32,  0, stream>>>(wsf);
  sae_zeroP  <<<2048, 256, 0, stream>>>(P);
  sae_cvt    <<<4096, 256, 0, stream>>>(Wenc, Wdec, wencTg, wdecTg);
  sae_rownorm<<<64,   256, 0, stream>>>(x, &wsf[0]);
  sae_scale  <<<1,    32,  0, stream>>>(&wsf[0], ravg, wsf);
  sae_wdnorm <<<512,  256, 0, stream>>>(Wdec, wdn);
  sae_bounds <<<1,    256, 0, stream>>>(seg, bnd);
  dim3 g1(BATCH / MT, NG);
  sae_phase1 <<<g1, 256, SMEM_BYTES, stream>>>(x, wencTg, benc, wdecTg,
                                               wdn, bnd, wsf, accum, P);
  sae_err    <<<BATCH, 256, 0, stream>>>(x, bdec, wsf, P, accum);
  sae_final  <<<1,    32,  0, stream>>>(accum, bw, l1sc, out);
}
